// VQVariationalAutoencoder_79551384256863
// MI455X (gfx1250) — compile-verified, matrix operand order checked by reading, not run on hardware
//
#include <hip/hip_runtime.h>
#include <hip/hip_bf16.h>

typedef _Float16 h8   __attribute__((ext_vector_type(8)));
typedef _Float16 v16h __attribute__((ext_vector_type(16)));
typedef float    v8f  __attribute__((ext_vector_type(8)));
typedef int      b128v __attribute__((vector_size(16)));   // pointee type of async builtins

#define BATCH 32
#define C1    128                 // conv1 out channels
#define DLAT  64                  // latent dims
#define KVEC  512                 // codebook size
#define HS    128                 // h / g spatial
#define HP    130                 // padded h / g spatial (1-px zero halo)
#define ZS    64                  // z spatial
#define ZP    66                  // padded idx spatial
#define IS    256                 // image spatial
#define NPIX  (BATCH*ZS*ZS)       // 131072 latent pixels
#define K2    (C1*16)             // conv2 GEMM depth = 2048
#define BSTR  40                  // LDS B-row stride in halves (80B: 16 lanes cover all 64 banks)

#if defined(__HIP_DEVICE_COMPILE__) && __has_builtin(__builtin_amdgcn_global_load_async_to_lds_b128)
#define HAVE_ASYNC 1
#else
#define HAVE_ASYNC 0
#endif

#if HAVE_ASYNC
typedef __attribute__((address_space(1))) b128v g_b128v;   // global (AS1)
typedef __attribute__((address_space(3))) b128v l_b128v;   // LDS (AS3)
#endif

static __device__ inline void cp16_g2l(const _Float16* gp, _Float16* lp) {
#if HAVE_ASYNC
  __builtin_amdgcn_global_load_async_to_lds_b128(
      (g_b128v*)(unsigned long long)gp,
      (l_b128v*)(unsigned int)(unsigned long long)lp,
      0, 0);
#else
  *(h8*)lp = *(const h8*)gp;
#endif
}
static __device__ inline void wait_async() {
#if HAVE_ASYNC
#if __has_builtin(__builtin_amdgcn_s_wait_asynccnt)
  __builtin_amdgcn_s_wait_asynccnt(0);
#else
  asm volatile("s_wait_asynccnt 0x0" ::: "memory");
#endif
#endif
}

static __device__ inline v16h cat8(h8 lo, h8 hi) {
  v16h r;
#pragma unroll
  for (int i = 0; i < 8; ++i) { r[i] = lo[i]; r[i + 8] = hi[i]; }
  return r;
}
static __device__ inline v8f wmma_f16(v16h a, v16h b, v8f c) {
  return __builtin_amdgcn_wmma_f32_16x16x32_f16(false, a, false, b, (short)0, c,
                                                false, false);
}

// ---------------- weight / codebook packing ----------------
__global__ void k_pack_enc2(const float* __restrict__ w, _Float16* __restrict__ B) {
  int t = blockIdx.x * blockDim.x + threadIdx.x;        // 64*2048
  int n = t >> 11, k = t & 2047;
  int tap = k >> 7, c = k & 127;
  B[t] = (_Float16)w[((size_t)n * C1 + c) * 16 + tap];
}
__global__ void k_pack_dec2(const float* __restrict__ w, _Float16* __restrict__ B) {
  int t = blockIdx.x * blockDim.x + threadIdx.x;        // 4*128*256
  int pc = t >> 15, rem = t & 32767;
  int n = rem >> 8, k = rem & 255;
  int tp = k >> 6, ci = k & 63;
  int py = pc >> 1, px = pc & 1;
  int ky = (py ? 0 : 1) + 2 * (tp >> 1);
  int kx = (px ? 0 : 1) + 2 * (tp & 1);
  B[t] = (_Float16)w[(((size_t)ci * C1 + n) * 4 + ky) * 4 + kx];
}
__global__ void k_pack_dec1(const float* __restrict__ w, _Float16* __restrict__ W) {
  int t = blockIdx.x * blockDim.x + threadIdx.x;        // 2048
  int pc = t >> 9, k = t & 511;
  int tp = k >> 7, ci = k & 127;
  int py = pc >> 1, px = pc & 1;
  int ky = (py ? 0 : 1) + 2 * (tp >> 1);
  int kx = (px ? 0 : 1) + 2 * (tp & 1);
  W[t] = (_Float16)w[(size_t)ci * 16 + ky * 4 + kx];
}
__global__ void k_pack_cb(const float* __restrict__ cb, _Float16* __restrict__ cbh,
                          float* __restrict__ cnorm, float* __restrict__ lossacc) {
  int k = blockIdx.x * blockDim.x + threadIdx.x;        // 512
  if (k == 0) {
    *lossacc = 0.f;
    for (int d = 0; d < DLAT; ++d) cbh[(size_t)KVEC * DLAT + d] = (_Float16)0.f; // zero row 512
  }
  float s = 0.f;
  for (int d = 0; d < DLAT; ++d) {
    float v = cb[(size_t)k * DLAT + d];
    cbh[(size_t)k * DLAT + d] = (_Float16)v;
    s += v * v;
  }
  cnorm[k] = s;
}

// ---------------- halo / border initialization (branchless gathers later) -------------
__global__ void k_zero_halo(_Float16* __restrict__ hg) {
  int t = blockIdx.x * blockDim.x + threadIdx.x;        // 32*130*130
  if (t >= BATCH * HP * HP) return;
  int r = t % (HP * HP);
  int y = r / HP, x = r % HP;
  if (y == 0 || y == HP - 1 || x == 0 || x == HP - 1) {
    h8 zv = {};
    _Float16* p = hg + (size_t)t * C1;
#pragma unroll
    for (int i = 0; i < 16; ++i) *(h8*)(p + i * 8) = zv;
  }
}
__global__ void k_idx_border(int* __restrict__ ip) {
  int t = blockIdx.x * blockDim.x + threadIdx.x;        // 32*66*66
  if (t >= BATCH * ZP * ZP) return;
  int r = t % (ZP * ZP);
  int y = r / ZP, x = r % ZP;
  if (y == 0 || y == ZP - 1 || x == 0 || x == ZP - 1) ip[t] = KVEC; // -> zero codebook row
}

// ---------------- encoder conv1 (Cin=1): direct, LDS-staged taps ----------------
__global__ void k_conv1(const float* __restrict__ x, const float* __restrict__ w,
                        const float* __restrict__ bias, _Float16* __restrict__ h) {
  __shared__ float taps[16];
  const int b  = blockIdx.y;
  const int oy = blockIdx.x >> 7;
  const int ox = blockIdx.x & 127;
  const int co = threadIdx.x;                           // 0..127
  if (co < 16) {
    int ky = co >> 2, kx = co & 3;
    int iy = oy * 2 - 1 + ky, ix = ox * 2 - 1 + kx;
    float v = 0.f;
    if ((unsigned)iy < IS && (unsigned)ix < IS)
      v = x[((size_t)b * IS + iy) * IS + ix];
    taps[co] = v;
  }
  __syncthreads();
  float acc = bias[co];
#pragma unroll
  for (int t = 0; t < 16; ++t) acc += taps[t] * w[co * 16 + t];
  if (acc < 0.f) acc = 0.f;
  h[(((size_t)b * HP + (oy + 1)) * HP + (ox + 1)) * C1 + co] = (_Float16)acc;
}

// ------- encoder conv2: 4-wave implicit GEMM, B double-buffered via async->LDS -------
__global__ void __launch_bounds__(128) k_conv2(const _Float16* __restrict__ h,
                                               const _Float16* __restrict__ Benc,
                                               const float* __restrict__ bias,
                                               _Float16* __restrict__ z) {
  __shared__ _Float16 sB[2][DLAT * BSTR];
  const int tid = threadIdx.x;
  const int wave = tid >> 5, lane = tid & 31;
  const int m = lane & 15, half = lane >> 4, kbase = half * 8;
  const int mtile = blockIdx.x * 4 + wave;
  const int p  = mtile * 16 + m;
  const int b  = p >> 12;
  const int oy = (p >> 6) & 63;
  const int ox = p & 63;
  const _Float16* abase = h + (((size_t)b * HP + oy * 2) * HP + ox * 2) * C1;

  auto stageB = [&](int buf, int kk) {                  // 64 rows x 32 halves
#pragma unroll
    for (int r = 0; r < 2; ++r) {
      int s = tid + r * 128;                            // 256 16B segments
      int n = s >> 2, q = s & 3;
      cp16_g2l(Benc + (size_t)n * K2 + kk + q * 8, &sB[buf][n * BSTR + q * 8]);
    }
  };

  stageB(0, 0);
  wait_async();
  __syncthreads();

  v8f acc[4] = {};
  for (int kk = 0; kk < K2; kk += 32) {
    const int cur = (kk >> 5) & 1;
    if (kk + 32 < K2) stageB(cur ^ 1, kk + 32);
    h8 a[2];
#pragma unroll
    for (int j = 0; j < 2; ++j) {                       // branchless: halo is zero
      int k = kk + kbase + j * 16;
      int tap = k >> 7, c0 = k & 127;
      int ky = tap >> 2, kx = tap & 3;
      a[j] = *(const h8*)(abase + (ky * HP + kx) * C1 + c0);
    }
    v16h A = cat8(a[0], a[1]);
#pragma unroll
    for (int nt = 0; nt < 4; ++nt) {
      const _Float16* bp = &sB[cur][(nt * 16 + m) * BSTR + kbase];
      v16h Bv = cat8(*(const h8*)bp, *(const h8*)(bp + 16));
      acc[nt] = wmma_f16(A, Bv, acc[nt]);
    }
    wait_async();
    __syncthreads();
  }
#pragma unroll
  for (int nt = 0; nt < 4; ++nt) {
    int n = nt * 16 + m;
    float bv = bias[n];
#pragma unroll
    for (int v = 0; v < 8; ++v) {
      int row = v + 8 * half;
      float val = acc[nt][v] + bv;
      if (val < 0.f) val = 0.f;
      z[(size_t)(mtile * 16 + row) * DLAT + n] = (_Float16)val;
    }
  }
}

// ---------------- VQ: z.c^T via WMMA, argmin over 512 codes, loss fragment ----------------
__global__ void k_vq(const _Float16* __restrict__ z, const _Float16* __restrict__ cb,
                     const float* __restrict__ cnorm, int* __restrict__ idx_out,
                     int* __restrict__ idx_pad, float* __restrict__ lossacc) {
  const int lane  = threadIdx.x;
  const int mtile = blockIdx.x;
  const int m     = lane & 15;
  const int half  = lane >> 4;
  const int kbase = half * 8;
  const _Float16* zr = z + (size_t)(mtile * 16 + m) * DLAT;
  h8 c0 = *(const h8*)(zr + kbase);
  h8 c1 = *(const h8*)(zr + kbase + 16);
  h8 c2 = *(const h8*)(zr + kbase + 32);
  h8 c3 = *(const h8*)(zr + kbase + 48);
  v16h A0 = cat8(c0, c1), A1 = cat8(c2, c3);

  float part = 0.f;
#pragma unroll
  for (int i = 0; i < 8; ++i)
    part += (float)c0[i]*(float)c0[i] + (float)c1[i]*(float)c1[i] +
            (float)c2[i]*(float)c2[i] + (float)c3[i]*(float)c3[i];
  float znorm = part + __shfl_xor(part, 16, 32);        // full ||z_row||^2

  float best[8]; int bidx[8];
#pragma unroll
  for (int v = 0; v < 8; ++v) { best[v] = 3.4e38f; bidx[v] = 0; }

  for (int nt = 0; nt < KVEC / 16; ++nt) {
    int n = nt * 16 + m;
    const _Float16* cr = cb + (size_t)n * DLAT;
    v16h B0 = cat8(*(const h8*)(cr + kbase), *(const h8*)(cr + kbase + 16));
    v16h B1 = cat8(*(const h8*)(cr + kbase + 32), *(const h8*)(cr + kbase + 48));
    v8f acc = {};
    acc = wmma_f16(A0, B0, acc);
    acc = wmma_f16(A1, B1, acc);
    float cn = cnorm[n];
#pragma unroll
    for (int v = 0; v < 8; ++v) {
      float s = cn - 2.f * acc[v];                      // ||c||^2 - 2 z.c
      if (s < best[v]) { best[v] = s; bidx[v] = n; }
    }
  }
#pragma unroll
  for (int mask = 1; mask <= 8; mask <<= 1) {
#pragma unroll
    for (int v = 0; v < 8; ++v) {
      float ob = __shfl_xor(best[v], mask, 32);
      int   oi = __shfl_xor(bidx[v], mask, 32);
      if (ob < best[v] || (ob == best[v] && oi < bidx[v])) { best[v] = ob; bidx[v] = oi; }
    }
  }
  float rowz[8];
#pragma unroll
  for (int v = 0; v < 8; ++v) rowz[v] = __shfl(znorm, v + 8 * half, 32);
  if (m == 0) {
    float s = 0.f;
#pragma unroll
    for (int v = 0; v < 8; ++v) {
      int row = v + 8 * half;
      int pr = mtile * 16 + row;
      idx_out[pr] = bidx[v];
      int bb = pr >> 12, zy = (pr >> 6) & 63, zx = pr & 63;
      idx_pad[((size_t)bb * ZP + (zy + 1)) * ZP + (zx + 1)] = bidx[v];
      s += best[v] + rowz[v];
    }
    atomicAdd(lossacc, s);
  }
}

__global__ void k_loss_final(const float* __restrict__ lossacc, float* __restrict__ out) {
  out[0] = lossacc[0] * 1.25f / ((float)NPIX * (float)DLAT);
}

// ------- decoder convT2: 4-wave parity-class GEMM, B double-buffered via async->LDS -------
__global__ void __launch_bounds__(128) k_convT2(const int* __restrict__ idxp,
                                                const _Float16* __restrict__ cb,
                                                const _Float16* __restrict__ Bdec,
                                                const float* __restrict__ bias,
                                                _Float16* __restrict__ g) {
  __shared__ _Float16 sB[2][C1 * BSTR];
  const int tid = threadIdx.x;
  const int wave = tid >> 5, lane = tid & 31;
  const int m = lane & 15, half = lane >> 4, kbase = half * 8;
  const int px = blockIdx.x;                            // 0..1
  const int oy = blockIdx.y, b = blockIdx.z;
  const int py = oy & 1, pc = py * 2 + px;
  const int chunk = wave;                               // ox2 range [chunk*16, +16)
  const int ox = (chunk * 16 + m) * 2 + px;
  const _Float16* Bsrc = Bdec + (size_t)pc * C1 * 256;

  auto stageB = [&](int buf, int kk) {                  // 128 rows x 32 halves
#pragma unroll
    for (int r = 0; r < 4; ++r) {
      int s = tid + r * 128;                            // 512 16B segments
      int n = s >> 2, q = s & 3;
      cp16_g2l(Bsrc + (size_t)n * 256 + kk + q * 8, &sB[buf][n * BSTR + q * 8]);
    }
  };

  stageB(0, 0);
  wait_async();
  __syncthreads();

  v8f acc[8] = {};
  for (int kk = 0; kk < 256; kk += 32) {
    const int cur = (kk >> 5) & 1;
    if (kk + 32 < 256) stageB(cur ^ 1, kk + 32);
    h8 a[2];
#pragma unroll
    for (int j = 0; j < 2; ++j) {                       // branchless via padded idx + zero row
      int k = kk + kbase + j * 16;
      int t = k >> 6, ci0 = k & 63;
      int ky = (py ? 0 : 1) + 2 * (t >> 1);
      int kx = (px ? 0 : 1) + 2 * (t & 1);
      int hy1 = ((oy + 1 - ky) >> 1) + 1;
      int hx1 = ((ox + 1 - kx) >> 1) + 1;
      int iv = idxp[((size_t)b * ZP + hy1) * ZP + hx1];
      a[j] = *(const h8*)(cb + (size_t)iv * DLAT + ci0);
    }
    v16h A = cat8(a[0], a[1]);
#pragma unroll
    for (int nt = 0; nt < 8; ++nt) {
      const _Float16* bp = &sB[cur][(nt * 16 + m) * BSTR + kbase];
      v16h Bv = cat8(*(const h8*)bp, *(const h8*)(bp + 16));
      acc[nt] = wmma_f16(A, Bv, acc[nt]);
    }
    wait_async();
    __syncthreads();
  }
#pragma unroll
  for (int nt = 0; nt < 8; ++nt) {
    int n = nt * 16 + m;
    float bv = bias[n];
#pragma unroll
    for (int v = 0; v < 8; ++v) {
      int row = v + 8 * half;
      int oxr = (chunk * 16 + row) * 2 + px;
      float val = acc[nt][v] + bv;
      if (val < 0.f) val = 0.f;
      g[(((size_t)b * HP + (oy + 1)) * HP + (oxr + 1)) * C1 + n] = (_Float16)val;
    }
  }
}

// ---------------- decoder convT1 (Cout=1): branchless vector dot + sigmoid ----------------
__global__ void k_convT1(const _Float16* __restrict__ g, const _Float16* __restrict__ w,
                         const float* __restrict__ bias, float* __restrict__ out) {
  int gid = blockIdx.x * blockDim.x + threadIdx.x;      // 32*256*256
  int b = gid >> 16;
  int oy = (gid >> 8) & 255;
  int ox = gid & 255;
  int py = oy & 1, px = ox & 1, pc = py * 2 + px;
  float acc = bias[0];
#pragma unroll
  for (int t = 0; t < 4; ++t) {
    int ky = (py ? 0 : 1) + 2 * (t >> 1);
    int kx = (px ? 0 : 1) + 2 * (t & 1);
    int gy1 = ((oy + 1 - ky) >> 1) + 1;                 // in [0,129], halo is zero
    int gx1 = ((ox + 1 - kx) >> 1) + 1;
    const _Float16* gr = g + (((size_t)b * HP + gy1) * HP + gx1) * C1;
    const _Float16* wr = w + pc * 512 + t * 128;
    for (int c8 = 0; c8 < 16; ++c8) {
      h8 gv = *(const h8*)(gr + c8 * 8);
      h8 wv = *(const h8*)(wr + c8 * 8);
#pragma unroll
      for (int e = 0; e < 8; ++e) acc += (float)gv[e] * (float)wv[e];
    }
  }
  out[gid] = 1.f / (1.f + __expf(-acc));
}

// ---------------- launch ----------------
extern "C" void kernel_launch(void* const* d_in, const int* in_sizes, int n_in,
                              void* d_out, int out_size, void* d_ws, size_t ws_size,
                              hipStream_t stream) {
  (void)in_sizes; (void)n_in; (void)out_size; (void)ws_size;
  const float* x        = (const float*)d_in[0];
  const float* enc_w1   = (const float*)d_in[1];
  const float* enc_b1   = (const float*)d_in[2];
  const float* enc_w2   = (const float*)d_in[3];
  const float* enc_b2   = (const float*)d_in[4];
  const float* codebook = (const float*)d_in[5];
  const float* dec_w2   = (const float*)d_in[6];
  const float* dec_b2   = (const float*)d_in[7];
  const float* dec_w1   = (const float*)d_in[8];
  const float* dec_b1   = (const float*)d_in[9];

  char* ws = (char*)d_ws;
  size_t off = 0;
  auto carve = [&](size_t bytes) {
    char* p = ws + off; off += (bytes + 255) & ~(size_t)255; return p;
  };
  _Float16* hg    = (_Float16*)carve((size_t)BATCH * HP * HP * C1 * 2); // h, reused as g
  _Float16* zbuf  = (_Float16*)carve((size_t)NPIX * DLAT * 2);
  _Float16* Benc  = (_Float16*)carve((size_t)DLAT * K2 * 2);
  _Float16* Bdec  = (_Float16*)carve((size_t)4 * C1 * 256 * 2);
  _Float16* Wd1   = (_Float16*)carve((size_t)4 * 512 * 2);
  _Float16* cbh   = (_Float16*)carve((size_t)(KVEC + 1) * DLAT * 2);    // +1 zero row
  float*    cnorm = (float*)carve((size_t)KVEC * 4);
  float*    lacc  = (float*)carve(256);
  int*      idpad = (int*)carve((size_t)BATCH * ZP * ZP * 4);

  float* outf = (float*)d_out;          // [0] = vq_loss
  int*   idxp = (int*)d_out + 1;        // [1 .. NPIX] = indices (int32 bits)
  float* xrec = outf + 1 + NPIX;        // [1+NPIX ..] = x_recon

  k_pack_enc2<<<DLAT * K2 / 256, 256, 0, stream>>>(enc_w2, Benc);
  k_pack_dec2<<<4 * C1 * 256 / 256, 256, 0, stream>>>(dec_w2, Bdec);
  k_pack_dec1<<<2048 / 256, 256, 0, stream>>>(dec_w1, Wd1);
  k_pack_cb<<<KVEC / 64, 64, 0, stream>>>(codebook, cbh, cnorm, lacc);
  k_zero_halo<<<(BATCH * HP * HP + 255) / 256, 256, 0, stream>>>(hg);
  k_idx_border<<<(BATCH * ZP * ZP + 255) / 256, 256, 0, stream>>>(idpad);
  k_conv1<<<dim3(HS * HS, BATCH), C1, 0, stream>>>(x, enc_w1, enc_b1, hg);
  k_conv2<<<NPIX / 64, 128, 0, stream>>>(hg, Benc, enc_b2, zbuf);
  k_vq<<<NPIX / 16, 32, 0, stream>>>(zbuf, cbh, cnorm, idxp, idpad, lacc);
  k_loss_final<<<1, 1, 0, stream>>>(lacc, outf);
  k_convT2<<<dim3(2, HS, BATCH), 128, 0, stream>>>(idpad, cbh, Bdec, dec_b2, hg);
  k_convT1<<<BATCH * IS * IS / 256, 256, 0, stream>>>(hg, Wd1, dec_b1, xrec);
}
